// ScSeTransformerBlock_14336600834050
// MI455X (gfx1250) — matrix operand order repaired and checked by reading, not compiled, not run on hardware
//
#include <hip/hip_runtime.h>
#include <hip/hip_bf16.h>
#include <math.h>
#include <stdint.h>

#define USE_ASYNC_LDS 1

// ---------------- problem constants ----------------
#define BATCH 4
#define CH    128
#define HH    192
#define WID   192
#define HEADS 4
#define HD    32
#define WIN_H 4
#define WIN_W 16
#define NWY   (HH / WIN_H)        // 48
#define NWX   (WID / WIN_W)       // 12
#define NWIN  (BATCH * NWY * NWX) // 2304
#define WF    (WID / 2 + 1)       // 97
#define HIDD  512
#define PI2   6.283185307179586f

static const size_t NPIX = (size_t)BATCH * HH * WID;      // 147456
static const size_t NPF  = (size_t)BATCH * HH * WF;       // 74496
static const size_t NF   = (size_t)BATCH * CH * HH * WF;  // 9,535,488

// ---------------- WMMA types ----------------
typedef __attribute__((ext_vector_type(16))) __bf16 v16bf;
typedef __attribute__((ext_vector_type(8)))  float  v8f;

union Frag16 {
  v16bf v;
  unsigned u[8];
  unsigned short s[16];
};

__device__ __forceinline__ unsigned short f2bf(float f) {
  union { float f; unsigned u; } x; x.f = f;
  unsigned r = x.u + 0x7FFFu + ((x.u >> 16) & 1u);  // round-to-nearest-even
  return (unsigned short)(r >> 16);
}
__device__ __forceinline__ float bf2f(unsigned short h) {
  union { unsigned u; float f; } x; x.u = ((unsigned)h) << 16; return x.f;
}
__device__ __forceinline__ float gelu_exact(float v) {
  return 0.5f * v * (1.0f + erff(v * 0.70710678118654752f));
}
__device__ __forceinline__ void wait_asynccnt0() {
#if __has_builtin(__builtin_amdgcn_s_wait_asynccnt)
  __builtin_amdgcn_s_wait_asynccnt(0);
#else
  asm volatile("s_wait_asynccnt 0x0" ::: "memory");
#endif
}

// ---------------- elementwise / setup kernels ----------------
// pack weight W[K x O] fp32 into K-pair-interleaved bf16 dwords:
// Wp[(k/2)*O + n] = { lo16 = bf16(W[k][n]), hi16 = bf16(W[k+1][n]) }
__global__ void k_pack_w_pairs(const float* __restrict__ w, unsigned* __restrict__ wp,
                               int K, int O) {
  size_t t = (size_t)blockIdx.x * blockDim.x + threadIdx.x;
  size_t total = (size_t)(K / 2) * O;
  if (t >= total) return;
  int n = (int)(t % O);
  int p = (int)(t / O);
  unsigned lo = f2bf(w[(size_t)(2 * p) * O + n]);
  unsigned hi = f2bf(w[(size_t)(2 * p + 1) * O + n]);
  wp[t] = lo | (hi << 16);
}

// depthwise 3x3x3 pos conv (singleton depth -> only center depth slice hits
// real data), residual-added; NCHW input -> NHWC fp32 output (residual stream)
__global__ void k_posconv(const float* __restrict__ x, const float* __restrict__ pw,
                          const float* __restrict__ pb, float* __restrict__ xt) {
  size_t t = (size_t)blockIdx.x * blockDim.x + threadIdx.x;
  if (t >= NPIX * CH) return;
  int c = (int)(t % CH);
  size_t p = t / CH;
  int w = (int)(p % WID);
  int h = (int)((p / WID) % HH);
  int b = (int)(p / ((size_t)WID * HH));
  const float* xc = x + ((size_t)b * CH + c) * HH * WID;
  const float* wc = pw + c * 27 + 9;  // center depth slice of (1,3,3,3)
  float acc = pb[c];
#pragma unroll
  for (int ky = 0; ky < 3; ++ky) {
    int hy = h + ky - 1;
    if (hy < 0 || hy >= HH) continue;
#pragma unroll
    for (int kx = 0; kx < 3; ++kx) {
      int wx = w + kx - 1;
      if (wx < 0 || wx >= WID) continue;
      acc += xc[(size_t)hy * WID + wx] * wc[ky * 3 + kx];
    }
  }
  xt[t] = xc[(size_t)h * WID + w] + acc;
}

// per-pixel LayerNorm over C=128, fp32 NHWC -> bf16 NHWC
__global__ void k_layernorm(const float* __restrict__ xt, const float* __restrict__ g,
                            const float* __restrict__ bta, unsigned short* __restrict__ out) {
  size_t p = (size_t)blockIdx.x * blockDim.x + threadIdx.x;
  if (p >= NPIX) return;
  const float* xp = xt + p * CH;
  float m = 0.f;
  for (int c = 0; c < CH; ++c) m += xp[c];
  m *= (1.0f / CH);
  float v = 0.f;
  for (int c = 0; c < CH; ++c) { float d = xp[c] - m; v += d * d; }
  v *= (1.0f / CH);
  float inv = rsqrtf(v + 1e-5f);
  unsigned short* op = out + p * CH;
  for (int c = 0; c < CH; ++c) op[c] = f2bf((xp[c] - m) * inv * g[c] + bta[c]);
}

// ---------------- generic WMMA GEMM ----------------
// Out[p, o] = epilogue( sum_k A[p,k] * Wp[k,o] + bias[o] )
// A: [P x K] bf16. Wp: K-pair packed bf16 dwords [(K/2) x O].
// One wave -> one 16x16 tile; 4 waves/block cover 64 output channels.
// The wave's 16xK A tile (contiguous 32*K bytes) is staged into per-wave LDS
// with GLOBAL_LOAD_ASYNC_TO_LDS_B128 (ASYNCcnt), then fragments read via DS.
// mode: 0 = bf16 store of (acc+bias)*scale
//       1 = bf16 store of GELU(acc+bias)
//       2 = bf16 store of LeakyReLU_{0.01}(acc+bias)
//       3 = fp32 outf[p*O+o] += acc+bias   (residual add)
//       4 = fp32 store
__global__ void k_gemm_wmma(const unsigned short* __restrict__ A,
                            const unsigned* __restrict__ Wp,
                            const float* __restrict__ bias,
                            unsigned short* __restrict__ outb,
                            float* __restrict__ outf,
                            int K, int O, float scale, int mode) {
  extern __shared__ char smem_dyn[];
  int lane = threadIdx.x & 31;
  int wave = threadIdx.x >> 5;
  size_t p0 = (size_t)blockIdx.x * 16;
  int n0 = (blockIdx.y * 4 + wave) * 16;
  int nlo = lane & 15, half = lane >> 4;

  char* aw = smem_dyn + (size_t)wave * K * 32;        // 16 rows * K * 2 bytes
#if USE_ASYNC_LDS
  {
    const char* gsrc = (const char*)(A + p0 * (size_t)K);  // 16 contiguous rows
    unsigned ldsbase = (unsigned)(uintptr_t)aw;
    int tilebytes = K * 32;
    for (int off = lane * 16; off < tilebytes; off += 32 * 16) {
      unsigned long long ga = (unsigned long long)(uintptr_t)(gsrc + off);
      unsigned la = ldsbase + (unsigned)off;
      asm volatile("global_load_async_to_lds_b128 %0, %1, off"
                   :: "v"(la), "v"(ga) : "memory");
    }
    wait_asynccnt0();
  }
#else
  {
    const unsigned* gsrc = (const unsigned*)(A + p0 * (size_t)K);
    unsigned* dst = (unsigned*)aw;
    int tiledw = K * 8;  // 16*K*2 bytes / 4
    for (int i = lane; i < tiledw; i += 32) dst[i] = gsrc[i];
  }
#endif

  const unsigned* Arow = (const unsigned*)(aw + (size_t)nlo * K * 2);
  v8f acc = {0.f, 0.f, 0.f, 0.f, 0.f, 0.f, 0.f, 0.f};
  for (int k0 = 0; k0 < K; k0 += 32) {
    Frag16 a, b;
    int base = (k0 >> 1) + half * 8;
#pragma unroll
    for (int r = 0; r < 4; ++r) {
      a.u[r]     = Arow[base + r];      // K = half*16 + 2r
      a.u[4 + r] = Arow[base + 4 + r];  // K = half*16 + 8 + 2r
    }
    // B fragment: pair index for K value (k0 + half*16 + 2r) is k0/2 + half*8 + r
    const unsigned* Bp = Wp + (size_t)((k0 >> 1) + half * 8) * O + n0 + nlo;
    if (k0 + 32 < K) __builtin_prefetch(Bp + (size_t)16 * O, 0, 1);
#pragma unroll
    for (int r = 0; r < 8; ++r) b.u[r] = Bp[(size_t)r * O];
    acc = __builtin_amdgcn_wmma_f32_16x16x32_bf16(false, a.v, false, b.v,
                                                  (short)0, acc, false, false);
  }
  float bv = bias[n0 + nlo];
#pragma unroll
  for (int r = 0; r < 8; ++r) {
    float v = acc[r] + bv;
    size_t off = (p0 + half * 8 + r) * (size_t)O + n0 + nlo;
    if (mode == 0)      outb[off] = f2bf(v * scale);
    else if (mode == 1) outb[off] = f2bf(gelu_exact(v));
    else if (mode == 2) outb[off] = f2bf(v >= 0.f ? v : 0.01f * v);
    else if (mode == 3) outf[off] += v;
    else                outf[off] = v;
  }
}

// ---------------- fused window attention ----------------
// One wave per (window, head): S = (q*scale) K^T + relpos -> softmax -> P V.
// q: [pix][128] bf16 (pre-scaled), kv: [pix][256] bf16 (k then v), ao: [pix][128] bf16
__global__ void k_win_attn(const unsigned short* __restrict__ q,
                           const unsigned short* __restrict__ kv,
                           const float* __restrict__ rpb,
                           unsigned short* __restrict__ ao) {
  __shared__ float S[64 * 64];
  __shared__ unsigned short Pb[64 * 64];
  int lane = threadIdx.x;
  int head = blockIdx.y;
  int wid = blockIdx.x;
  int b = wid / (NWY * NWX);
  int rem = wid % (NWY * NWX);
  int wy = rem / NWX, wx = rem % NWX;
  int nlo = lane & 15, half = lane >> 4;
  auto pix = [&](int e) -> size_t {
    int hy = wy * WIN_H + (e >> 4);
    int hx = wx * WIN_W + (e & 15);
    return ((size_t)b * HH + hy) * WID + hx;
  };

  // ---- S = q k^T (+ relative position bias) ----
  for (int mt = 0; mt < 4; ++mt) {
    Frag16 a;
    const unsigned* Aq = (const unsigned*)(q + pix(mt * 16 + nlo) * CH + head * HD);
#pragma unroll
    for (int r = 0; r < 4; ++r) { a.u[r] = Aq[half * 8 + r]; a.u[4 + r] = Aq[half * 8 + 4 + r]; }
    for (int nt = 0; nt < 4; ++nt) {
      Frag16 bf;  // B[k][n] = K[element n][feature k]; features contiguous -> dword loads
      const unsigned* Kc = (const unsigned*)(kv + pix(nt * 16 + nlo) * (2 * CH) + head * HD);
#pragma unroll
      for (int r = 0; r < 8; ++r) bf.u[r] = Kc[half * 8 + r];
      v8f acc = {0.f, 0.f, 0.f, 0.f, 0.f, 0.f, 0.f, 0.f};
      acc = __builtin_amdgcn_wmma_f32_16x16x32_bf16(false, a.v, false, bf.v,
                                                    (short)0, acc, false, false);
#pragma unroll
      for (int r = 0; r < 8; ++r) {
        int row = mt * 16 + half * 8 + r;
        int col = nt * 16 + nlo;
        int dh = (row >> 4) - (col >> 4) + (WIN_H - 1);
        int dw = (row & 15) - (col & 15) + (WIN_W - 1);
        S[row * 64 + col] = acc[r] + rpb[(dh * (2 * WIN_W - 1) + dw) * HEADS + head];
      }
    }
  }
  __syncthreads();

  // ---- row softmax (64 rows over 32 lanes), write probs as bf16 ----
  for (int row = lane; row < 64; row += 32) {
    float* sr = S + row * 64;
    float mx = sr[0];
    for (int c2 = 1; c2 < 64; ++c2) mx = fmaxf(mx, sr[c2]);
    float sum = 0.f;
    for (int c2 = 0; c2 < 64; ++c2) { float e = __expf(sr[c2] - mx); sr[c2] = e; sum += e; }
    float inv = 1.f / sum;
    for (int c2 = 0; c2 < 64; ++c2) Pb[row * 64 + c2] = f2bf(sr[c2] * inv);
  }
  __syncthreads();

  // ---- out = P @ V ----
  for (int mt = 0; mt < 4; ++mt) {
    for (int nt = 0; nt < 2; ++nt) {
      v8f acc = {0.f, 0.f, 0.f, 0.f, 0.f, 0.f, 0.f, 0.f};
      for (int kc = 0; kc < 2; ++kc) {
        Frag16 a;
        const unsigned* Ap = (const unsigned*)(Pb + (mt * 16 + nlo) * 64 + kc * 32);
#pragma unroll
        for (int r = 0; r < 4; ++r) { a.u[r] = Ap[half * 8 + r]; a.u[4 + r] = Ap[half * 8 + 4 + r]; }
        Frag16 bf;  // B[k][n] = V[element k][feature n]
        int kb = kc * 32 + half * 16;
#pragma unroll
        for (int r = 0; r < 8; ++r) {
          bf.s[2 * r]     = kv[pix(kb + 2 * r)     * (2 * CH) + CH + head * HD + nt * 16 + nlo];
          bf.s[2 * r + 1] = kv[pix(kb + 2 * r + 1) * (2 * CH) + CH + head * HD + nt * 16 + nlo];
        }
        acc = __builtin_amdgcn_wmma_f32_16x16x32_bf16(false, a.v, false, bf.v,
                                                      (short)0, acc, false, false);
      }
#pragma unroll
      for (int r = 0; r < 8; ++r)
        ao[pix(mt * 16 + half * 8 + r) * CH + head * HD + nt * 16 + nlo] = f2bf(acc[r]);
    }
  }
}

// ---------------- FFT branch (explicit DFTs) ----------------
__global__ void k_dft_w(const float* __restrict__ xt, float* __restrict__ Fr,
                        float* __restrict__ Fi) {
  size_t t = (size_t)blockIdx.x * blockDim.x + threadIdx.x;
  if (t >= NF) return;
  int kw = (int)(t % WF);
  int h  = (int)((t / WF) % HH);
  int c  = (int)((t / ((size_t)WF * HH)) % CH);
  int b  = (int)(t / ((size_t)WF * HH * CH));
  float w0 = -PI2 * (float)kw / (float)WID;
  const float* row = xt + ((size_t)b * HH + h) * WID * CH + c;
  float sr = 0.f, si = 0.f;
  for (int w = 0; w < WID; ++w) {
    float s, cc; __sincosf(w0 * (float)w, &s, &cc);
    float xv = row[(size_t)w * CH];
    sr += xv * cc; si += xv * s;
  }
  Fr[t] = sr; Fi[t] = si;
}

__global__ void k_dft_h(const float* __restrict__ Fr, const float* __restrict__ Fi,
                        float* __restrict__ Gr, float* __restrict__ Gi) {
  size_t t = (size_t)blockIdx.x * blockDim.x + threadIdx.x;
  if (t >= NF) return;
  int kw = (int)(t % WF);
  int kh = (int)((t / WF) % HH);
  int c  = (int)((t / ((size_t)WF * HH)) % CH);
  int b  = (int)(t / ((size_t)WF * HH * CH));
  float w0 = -PI2 * (float)kh / (float)HH;
  size_t base = ((size_t)b * CH + c) * HH * WF + kw;
  float sr = 0.f, si = 0.f;
  for (int h = 0; h < HH; ++h) {
    float s, cc; __sincosf(w0 * (float)h, &s, &cc);
    float fr = Fr[base + (size_t)h * WF], fi = Fi[base + (size_t)h * WF];
    sr += fr * cc - fi * s;
    si += fr * s + fi * cc;
  }
  Gr[t] = sr; Gi[t] = si;
}

__global__ void k_pack_ff(const float* __restrict__ Gr, const float* __restrict__ Gi,
                          unsigned short* __restrict__ ffb) {
  size_t t = (size_t)blockIdx.x * blockDim.x + threadIdx.x;
  if (t >= NPF * CH) return;
  int c = (int)(t % CH);
  size_t pf = t / CH;
  int kw = (int)(pf % WF);
  int kh = (int)((pf / WF) % HH);
  int b  = (int)(pf / ((size_t)WF * HH));
  size_t src = ((size_t)b * CH + c) * HH * WF + (size_t)kh * WF + kw;
  ffb[pf * (2 * CH) + c]      = f2bf(Gr[src]);
  ffb[pf * (2 * CH) + CH + c] = f2bf(Gi[src]);
}

__global__ void k_idft_h(const float* __restrict__ ffo, float* __restrict__ g2r,
                         float* __restrict__ g2i) {
  size_t t = (size_t)blockIdx.x * blockDim.x + threadIdx.x;
  if (t >= NF) return;
  int kw = (int)(t % WF);
  int h  = (int)((t / WF) % HH);
  int c  = (int)((t / ((size_t)WF * HH)) % CH);
  int b  = (int)(t / ((size_t)WF * HH * CH));
  float w0 = PI2 * (float)h / (float)HH;
  float sr = 0.f, si = 0.f;
  for (int kh = 0; kh < HH; ++kh) {
    float s, cc; __sincosf(w0 * (float)kh, &s, &cc);
    size_t pf = ((size_t)b * HH + kh) * WF + kw;
    float fr = ffo[pf * (2 * CH) + c];
    float fi = ffo[pf * (2 * CH) + CH + c];
    sr += fr * cc - fi * s;
    si += fr * s + fi * cc;
  }
  g2r[t] = sr; g2i[t] = si;
}

__global__ void k_idft_w_add(const float* __restrict__ g2r, const float* __restrict__ g2i,
                             float* __restrict__ xt) {
  size_t t = (size_t)blockIdx.x * blockDim.x + threadIdx.x;
  if (t >= NPIX * CH) return;
  int c = (int)(t % CH);
  size_t p = t / CH;
  int w = (int)(p % WID);
  int h = (int)((p / WID) % HH);
  int b = (int)(p / ((size_t)WID * HH));
  float w0 = PI2 * (float)w / (float)WID;
  size_t base = ((size_t)b * CH + c) * HH * WF + (size_t)h * WF;
  float acc = 0.f;
  for (int kw = 0; kw < WF; ++kw) {
    float s, cc; __sincosf(w0 * (float)kw, &s, &cc);
    float wg = (kw == 0 || kw == WF - 1) ? 1.f : 2.f;
    acc += wg * (g2r[base + kw] * cc - g2i[base + kw] * s);
  }
  xt[t] += acc * (1.f / ((float)HH * (float)WID));
}

// ---------------- LeFF depthwise conv ----------------
__global__ void k_dwconv_gelu(const unsigned short* __restrict__ h1,
                              const float* __restrict__ dww, const float* __restrict__ dwb,
                              unsigned short* __restrict__ h2) {
  size_t t = (size_t)blockIdx.x * blockDim.x + threadIdx.x;
  if (t >= NPIX * HIDD) return;
  int ch = (int)(t % HIDD);
  size_t p = t / HIDD;
  int w = (int)(p % WID);
  int h = (int)((p / WID) % HH);
  int b = (int)(p / ((size_t)WID * HH));
  float acc = dwb[ch];
  const float* wc = dww + ch * 9;
#pragma unroll
  for (int ky = 0; ky < 3; ++ky) {
    int hy = h + ky - 1;
    if (hy < 0 || hy >= HH) continue;
#pragma unroll
    for (int kx = 0; kx < 3; ++kx) {
      int wx = w + kx - 1;
      if (wx < 0 || wx >= WID) continue;
      acc += bf2f(h1[(((size_t)b * HH + hy) * WID + wx) * HIDD + ch]) * wc[ky * 3 + kx];
    }
  }
  h2[t] = f2bf(gelu_exact(acc));
}

// final NHWC fp32 -> NCHW fp32 output
__global__ void k_nhwc_to_nchw(const float* __restrict__ xt, float* __restrict__ out) {
  size_t t = (size_t)blockIdx.x * blockDim.x + threadIdx.x;
  if (t >= NPIX * CH) return;
  int c = (int)(t % CH);
  size_t p = t / CH;
  int w = (int)(p % WID);
  int h = (int)((p / WID) % HH);
  int b = (int)(p / ((size_t)WID * HH));
  out[(((size_t)b * CH + c) * HH + h) * WID + w] = xt[t];
}

// ---------------- host launcher ----------------
extern "C" void kernel_launch(void* const* d_in, const int* in_sizes, int n_in,
                              void* d_out, int out_size, void* d_ws, size_t ws_size,
                              hipStream_t stream) {
  (void)in_sizes; (void)n_in; (void)out_size; (void)ws_size;
  const float* x      = (const float*)d_in[0];
  const float* pos_w  = (const float*)d_in[1];
  const float* pos_b  = (const float*)d_in[2];
  const float* ln1_g  = (const float*)d_in[3];
  const float* ln1_b  = (const float*)d_in[4];
  const float* q_w    = (const float*)d_in[5];
  const float* q_b    = (const float*)d_in[6];
  const float* kv_w   = (const float*)d_in[7];
  const float* kv_b   = (const float*)d_in[8];
  const float* rpb    = (const float*)d_in[9];
  const float* proj_w = (const float*)d_in[10];
  const float* proj_b = (const float*)d_in[11];
  const float* fft1_w = (const float*)d_in[12];
  const float* fft1_b = (const float*)d_in[13];
  const float* fft2_w = (const float*)d_in[14];
  const float* fft2_b = (const float*)d_in[15];
  const float* ln2_g  = (const float*)d_in[16];
  const float* ln2_b  = (const float*)d_in[17];
  const float* l1_w   = (const float*)d_in[18];
  const float* l1_b   = (const float*)d_in[19];
  const float* dw_w   = (const float*)d_in[20];
  const float* dw_b   = (const float*)d_in[21];
  const float* l2_w   = (const float*)d_in[22];
  const float* l2_b   = (const float*)d_in[23];

  // ----- workspace arena -----
  char* ws = (char*)d_ws;
  const size_t SZ_XT = NPIX * CH * sizeof(float);
  const size_t SZ_BF = NPIX * CH * sizeof(unsigned short);
  float* xt = (float*)ws;
  unsigned short* xn  = (unsigned short*)(ws + SZ_XT);
  unsigned short* qb  = (unsigned short*)(ws + SZ_XT + SZ_BF);       // q  [P x 128]
  unsigned short* kvb = qb + NPIX * CH;                              // kv [P x 256]
  unsigned short* aob = kvb + NPIX * 2 * CH;                         // attn out
  unsigned short* h1b = qb;  // LeFF hidden reuses q/kv/ao (exactly NPIX*512 bf16)
  char* fftbase = ws + SZ_XT + SZ_BF + (size_t)NPIX * HIDD * sizeof(unsigned short);
  float* Fr = (float*)fftbase;
  float* Fi = Fr + NF;
  float* Gr = Fi + NF;
  float* Gi = Gr + NF;
  float* ffo = (float*)fftbase;           // fft2 output (2*NF floats)
  float* g2r = (float*)fftbase + 2 * NF;  // overlays Gr
  float* g2i = g2r + NF;                  // overlays Gi
  unsigned short* h2b = (unsigned short*)fftbase;
  unsigned short* ffb = (unsigned short*)(fftbase + 4 * NF * sizeof(float));
  unsigned short* ffh = ffb + NPF * 2 * CH;
  unsigned* wq  = (unsigned*)(ffh + NPF * 2 * CH);   // pair-packed bf16 weights
  unsigned* wkv = wq  + (CH / 2) * CH;
  unsigned* wpr = wkv + (CH / 2) * 2 * CH;
  unsigned* wf1 = wpr + (CH / 2) * CH;
  unsigned* wf2 = wf1 + CH * 2 * CH;       // (2CH/2)*2CH
  unsigned* wl1 = wf2 + CH * 2 * CH;
  unsigned* wl2 = wl1 + (CH / 2) * HIDD;

  auto pack = [&](const float* s, unsigned* d, int K, int O) {
    size_t n = (size_t)(K / 2) * O;
    k_pack_w_pairs<<<dim3((unsigned)((n + 255) / 256)), 256, 0, stream>>>(s, d, K, O);
  };
  pack(q_w,    wq,  CH,     CH);
  pack(kv_w,   wkv, CH,     2 * CH);
  pack(proj_w, wpr, CH,     CH);
  pack(fft1_w, wf1, 2 * CH, 2 * CH);
  pack(fft2_w, wf2, 2 * CH, 2 * CH);
  pack(l1_w,   wl1, CH,     HIDD);
  pack(l2_w,   wl2, HIDD,   CH);

  const unsigned EB = 256;
  // 1) positional conv + residual -> xt (NHWC fp32)
  k_posconv<<<dim3((unsigned)((NPIX * CH + EB - 1) / EB)), EB, 0, stream>>>(x, pos_w, pos_b, xt);
  // 2) LN1 -> xn bf16
  k_layernorm<<<dim3((unsigned)((NPIX + EB - 1) / EB)), EB, 0, stream>>>(xt, ln1_g, ln1_b, xn);
  // 3) q / kv projections (q pre-scaled by hd^-0.5)
  dim3 gq((unsigned)(NPIX / 16), CH / 64);
  k_gemm_wmma<<<gq, 128, (size_t)CH * 128, stream>>>(xn, wq, q_b, qb, nullptr, CH, CH,
                                                     0.17677669529663687f, 0);
  dim3 gkv((unsigned)(NPIX / 16), (2 * CH) / 64);
  k_gemm_wmma<<<gkv, 128, (size_t)CH * 128, stream>>>(xn, wkv, kv_b, kvb, nullptr,
                                                      CH, 2 * CH, 1.f, 0);
  // 4) fused window attention
  k_win_attn<<<dim3(NWIN, HEADS), 32, 0, stream>>>(qb, kvb, rpb, aob);
  // 5) output projection, residual-add into xt
  k_gemm_wmma<<<gq, 128, (size_t)CH * 128, stream>>>(aob, wpr, proj_b, nullptr, xt,
                                                     CH, CH, 1.f, 3);
  // 6) FFT branch
  k_dft_w<<<dim3((unsigned)((NF + EB - 1) / EB)), EB, 0, stream>>>(xt, Fr, Fi);
  k_dft_h<<<dim3((unsigned)((NF + EB - 1) / EB)), EB, 0, stream>>>(Fr, Fi, Gr, Gi);
  k_pack_ff<<<dim3((unsigned)((NPF * CH + EB - 1) / EB)), EB, 0, stream>>>(Gr, Gi, ffb);
  dim3 gf((unsigned)(NPF / 16), (2 * CH) / 64);
  k_gemm_wmma<<<gf, 128, (size_t)(2 * CH) * 128, stream>>>(ffb, wf1, fft1_b, ffh, nullptr,
                                                           2 * CH, 2 * CH, 1.f, 2);
  k_gemm_wmma<<<gf, 128, (size_t)(2 * CH) * 128, stream>>>(ffh, wf2, fft2_b, nullptr, ffo,
                                                           2 * CH, 2 * CH, 1.f, 4);
  k_idft_h<<<dim3((unsigned)((NF + EB - 1) / EB)), EB, 0, stream>>>(ffo, g2r, g2i);
  k_idft_w_add<<<dim3((unsigned)((NPIX * CH + EB - 1) / EB)), EB, 0, stream>>>(g2r, g2i, xt);
  // 7) LeFF
  k_layernorm<<<dim3((unsigned)((NPIX + EB - 1) / EB)), EB, 0, stream>>>(xt, ln2_g, ln2_b, xn);
  dim3 gl1((unsigned)(NPIX / 16), HIDD / 64);
  k_gemm_wmma<<<gl1, 128, (size_t)CH * 128, stream>>>(xn, wl1, l1_b, h1b, nullptr,
                                                      CH, HIDD, 1.f, 1);
  k_dwconv_gelu<<<dim3((unsigned)((NPIX * HIDD + EB - 1) / EB)), EB, 0, stream>>>(h1b, dw_w, dw_b, h2b);
  k_gemm_wmma<<<gq, 128, (size_t)HIDD * 128, stream>>>(h2b, wl2, l2_b, nullptr, xt,
                                                       HIDD, CH, 1.f, 3);
  // 8) NHWC -> NCHW output
  k_nhwc_to_nchw<<<dim3((unsigned)((NPIX * CH + EB - 1) / EB)), EB, 0, stream>>>(xt, (float*)d_out);
}
